// PAN_30846455120742
// MI455X (gfx1250) — matrix operand order repaired and checked by reading, not compiled, not run on hardware
//
#include <hip/hip_runtime.h>

// PANConv x2 + log_softmax for N=4096, E=65536, F=500, H=64, C=40.
// Dominant op: dense A@A (4096^3) via v_wmma_f32_16x16x32_f16 with
// double-buffered async Global->LDS tile staging (ASYNCcnt pipeline).
// A holds small integer edge counts -> f16 inputs exact, f32 accumulate exact.

typedef __attribute__((ext_vector_type(16))) _Float16 v16h;
typedef __attribute__((ext_vector_type(8)))  float    v8f;

#define NN 4096
#define FF 500
#define FP 512
#define HH 64
#define CC 40
#define EE 65536

#define BM 128
#define BN 64
#define BK 32

union FragU { unsigned int u[8]; v16h v; };

// Low 32 bits of a generic pointer into LDS == workgroup-relative LDS byte
// offset (ISA 10.2).
__device__ __forceinline__ unsigned lds_off32(const void* p) {
  return (unsigned)(unsigned long long)p;
}

// Per-lane 16B async copy global -> LDS (VDST = LDS byte-offset VGPR).
__device__ __forceinline__ void async_cp16(unsigned lds, unsigned long long gaddr) {
  asm volatile("global_load_async_to_lds_b128 %0, %1, off"
               :: "v"(lds), "v"(gaddr) : "memory");
}

__device__ __forceinline__ void wait_async0() {
  asm volatile("s_wait_asynccnt 0" ::: "memory");
}

// ---------------------------------------------------------------- utilities
__global__ void zero_f32(float* p, size_t n) {
  size_t i = (size_t)blockIdx.x * blockDim.x + threadIdx.x;
  size_t stride = (size_t)gridDim.x * blockDim.x;
  for (; i < n; i += stride) p[i] = 0.0f;
}

__global__ void scatter_edges(const long long* __restrict__ ei,
                              float* __restrict__ A, float* __restrict__ rowA,
                              int E, int N) {
  int e = blockIdx.x * blockDim.x + threadIdx.x;
  if (e >= E) return;
  int r = (int)ei[e];
  int c = (int)ei[(size_t)E + e];
  atomicAdd(&A[(size_t)r * N + c], 1.0f);
  atomicAdd(&rowA[r], 1.0f);
}

__global__ void convert_A(const float* __restrict__ A,
                          _Float16* __restrict__ Ah, _Float16* __restrict__ Ath, int N) {
  size_t idx = (size_t)blockIdx.x * blockDim.x + threadIdx.x;
  int i = (int)(idx / N), j = (int)(idx % N);
  float a = A[idx];
  Ah[idx] = (_Float16)a;
  Ath[(size_t)j * N + i] = (_Float16)a;
}

__global__ void rowsum_f32(const float* __restrict__ Mat, float* __restrict__ out, int N) {
  __shared__ float red[256];
  int i = blockIdx.x;
  int tid = threadIdx.x;
  float s = 0.0f;
  for (int j = tid; j < N; j += 256) s += Mat[(size_t)i * N + j];
  red[tid] = s;
  __syncthreads();
  for (int st = 128; st > 0; st >>= 1) {
    if (tid < st) red[tid] += red[tid + st];
    __syncthreads();
  }
  if (tid == 0) out[i] = red[0];
}

__global__ void compute_dis(const float* __restrict__ w, const float* __restrict__ rowA,
                            const float* __restrict__ rowA2, float* __restrict__ dis, int N) {
  int i = blockIdx.x * blockDim.x + threadIdx.x;
  if (i >= N) return;
  float c0 = w[0], c1 = c0 * w[1], c2 = c1 * w[2];
  float deg = c0 + c1 * rowA[i] + c2 * rowA2[i];
  dis[i] = (deg > 0.0f) ? rsqrtf(fmaxf(deg, 1e-30f)) : 0.0f;
}

__global__ void build_P(const float* __restrict__ w, const float* __restrict__ A,
                        const float* __restrict__ A2, _Float16* __restrict__ P, size_t total) {
  size_t idx = (size_t)blockIdx.x * blockDim.x + threadIdx.x;
  if (idx >= total) return;
  float c0 = w[0], c1 = c0 * w[1], c2 = c1 * w[2];
  P[idx] = (_Float16)(c1 * A[idx] + c2 * A2[idx]);
}

// xt[f*N + i] = dis[i] * x[i*F + f]  (f16, zero-padded for f in [F, Fp))
__global__ void build_xst(const float* __restrict__ x, const float* __restrict__ dis,
                          _Float16* __restrict__ xt, int N, int F, int Fp) {
  size_t idx = (size_t)blockIdx.x * blockDim.x + threadIdx.x;
  if (idx >= (size_t)Fp * N) return;
  int f = (int)(idx / N), i = (int)(idx % N);
  xt[idx] = (f < F) ? (_Float16)(dis[i] * x[(size_t)i * F + f]) : (_Float16)0.0f;
}

// out[i,o] = act( b[o] + sum_f W[o,f] * dis_i*(c0*dis_i*x[i,f] + y[i,f]) )
__global__ void linear_comb(const float* __restrict__ w, const float* __restrict__ dis,
                            const float* __restrict__ x, const float* __restrict__ y,
                            const float* __restrict__ W, const float* __restrict__ b,
                            float* __restrict__ out, int N, int F, int ldy, int Co, int relu) {
  int idx = blockIdx.x * blockDim.x + threadIdx.x;
  if (idx >= N * Co) return;
  int i = idx / Co, o = idx % Co;
  float c0 = w[0];
  float di = dis[i];
  float acc = b[o];
  for (int f = 0; f < F; ++f) {
    float g = di * (c0 * di * x[(size_t)i * F + f] + y[(size_t)i * ldy + f]);
    acc = fmaf(g, W[(size_t)o * F + f], acc);
  }
  if (relu) acc = fmaxf(acc, 0.0f);
  out[(size_t)i * Co + o] = acc;
}

__global__ void logsoftmax_rows(float* __restrict__ z, int N, int Cc) {
  int i = blockIdx.x * blockDim.x + threadIdx.x;
  if (i >= N) return;
  float* row = z + (size_t)i * Cc;
  float m = -1e30f;
  for (int c = 0; c < Cc; ++c) m = fmaxf(m, row[c]);
  float s = 0.0f;
  for (int c = 0; c < Cc; ++c) s += expf(row[c] - m);
  float ls = logf(s) + m;
  for (int c = 0; c < Cc; ++c) row[c] = row[c] - ls;
}

// ------------------------------------------------------ WMMA f16 tiled GEMM
// C[M x Nn] = Amat[M x K] * B[K x Nn], B supplied transposed (Bt[Nn x K]).
// M % 128 == 0, Nn % 64 == 0, K % 32 == 0.
// 128x64 block tile, 8 wave32s (4x2), each wave 2x2 16x16x32 WMMA tiles.
// Tiles staged with double-buffered GLOBAL_LOAD_ASYNC_TO_LDS_B128 (ASYNCcnt).
__global__ __launch_bounds__(256) void wmma_gemm(const _Float16* __restrict__ Amat,
                                                 const _Float16* __restrict__ Bt,
                                                 float* __restrict__ Cmat,
                                                 int M, int Nn, int K) {
  __shared__ __align__(16) _Float16 As[2][BM * BK];  // 2 x 8 KB
  __shared__ __align__(16) _Float16 Bs[2][BN * BK];  // 2 x 4 KB
  const int tid  = threadIdx.x;
  const int lane = tid & 31;
  const int wave = tid >> 5;
  const int wm   = wave >> 1;   // 0..3
  const int wn   = wave & 1;    // 0..1
  const int bm0  = blockIdx.y * BM;
  const int bn0  = blockIdx.x * BN;
  const int hi   = lane >> 4;
  const int l15  = lane & 15;
  const int niter = K / BK;

  // Issue one K-step's tiles into LDS buffer `buf` via async 16B copies.
  // A tile: 128 rows x 64B = 512 chunks (2 per thread); B tile: 64 rows x 64B.
  auto issue = [&](int buf, int k0) {
    unsigned aB = lds_off32(&As[buf][0]);
    unsigned bB = lds_off32(&Bs[buf][0]);
    #pragma unroll
    for (int t = 0; t < 2; ++t) {
      int chunk = tid + t * 256;
      int r = chunk >> 2, p = chunk & 3;
      async_cp16(aB + chunk * 16,
                 (unsigned long long)(Amat + (size_t)(bm0 + r) * K + k0) + (unsigned)(p * 16));
    }
    int r = tid >> 2, p = tid & 3;
    async_cp16(bB + tid * 16,
               (unsigned long long)(Bt + (size_t)(bn0 + r) * K + k0) + (unsigned)(p * 16));
  };

  v8f acc[2][2] = {};

  issue(0, 0);
  for (int it = 0; it < niter; ++it) {
    wait_async0();       // own async writes to buf[it&1] complete
    __syncthreads();     // publish buf[it&1]; all reads of buf[(it+1)&1] are done
    if (it + 1 < niter) issue((it + 1) & 1, (it + 1) * BK);

    const _Float16* Ac = &As[it & 1][0];
    const _Float16* Bc = &Bs[it & 1][0];

    // Fragments in contiguous k order, k = 16*hi + s, same for A and B (not the ISA table order):
    //  A 16x32: lanes 0-15 row m, K 0..15;         lanes 16-31 row m, K 16..31
    //  B 32x16: lanes 0-15 col n, K 0..15;         lanes 16-31 col n, K 16..31
    FragU a[2], b[2];
    #pragma unroll
    for (int i = 0; i < 2; ++i) {
      const unsigned int* Arow = (const unsigned int*)Ac + (wm * 32 + i * 16 + l15) * 16;
      #pragma unroll
      for (int v = 0; v < 8; ++v) {
        int kb = (hi ? 16 : 0) + (v << 1);
        a[i].u[v] = Arow[kb >> 1];
      }
    }
    #pragma unroll
    for (int j = 0; j < 2; ++j) {
      const unsigned int* Brow = (const unsigned int*)Bc + (wn * 32 + j * 16 + l15) * 16;
      int kof = hi ? 8 : 0;
      #pragma unroll
      for (int v = 0; v < 8; ++v) b[j].u[v] = Brow[kof + v];
    }

    #pragma unroll
    for (int i = 0; i < 2; ++i)
      #pragma unroll
      for (int j = 0; j < 2; ++j)
        acc[i][j] = __builtin_amdgcn_wmma_f32_16x16x32_f16(
            false, a[i].v, false, b[j].v, (short)0, acc[i][j], false, false);
  }

  // C/D layout: VGPR v -> row (hi?8:0)+v, lane&15 -> col.
  #pragma unroll
  for (int i = 0; i < 2; ++i) {
    #pragma unroll
    for (int j = 0; j < 2; ++j) {
      int r0 = bm0 + wm * 32 + i * 16 + (hi ? 8 : 0);
      int c0 = bn0 + wn * 32 + j * 16 + l15;
      #pragma unroll
      for (int v = 0; v < 8; ++v)
        Cmat[(size_t)(r0 + v) * Nn + c0] = acc[i][j][v];
    }
  }
}

// ---------------------------------------------------------------- launcher
extern "C" void kernel_launch(void* const* d_in, const int* in_sizes, int n_in,
                              void* d_out, int out_size, void* d_ws, size_t ws_size,
                              hipStream_t stream) {
  const float*     x  = (const float*)d_in[0];
  const long long* ei = (const long long*)d_in[1];
  const float*     w1 = (const float*)d_in[2];
  const float*     W1 = (const float*)d_in[3];
  const float*     b1 = (const float*)d_in[4];
  const float*     w2 = (const float*)d_in[5];
  const float*     W2 = (const float*)d_in[6];
  const float*     b2 = (const float*)d_in[7];
  float* out = (float*)d_out;

  char* ws = (char*)d_ws;
  size_t off = 0;
  auto alloc = [&](size_t bytes) -> void* {
    void* p = ws + off;
    off += (bytes + 255) & ~(size_t)255;
    return p;
  };
  float*    A32   = (float*)   alloc((size_t)NN * NN * 4);   // dense adjacency (counts)
  float*    A2    = (float*)   alloc((size_t)NN * NN * 4);   // A@A
  _Float16* Ah    = (_Float16*)alloc((size_t)NN * NN * 2);   // A f16 row-major
  _Float16* Ath   = (_Float16*)alloc((size_t)NN * NN * 2);   // A^T f16 row-major
  _Float16* Ph    = (_Float16*)alloc((size_t)NN * NN * 2);   // c1*A + c2*A2 (per layer)
  _Float16* xt    = (_Float16*)alloc((size_t)FP * NN * 2);   // (dis*x)^T f16
  float*    y     = (float*)   alloc((size_t)NN * FP * 4);   // P @ xs
  float*    h     = (float*)   alloc((size_t)NN * HH * 4);   // layer-1 output
  float*    rowA  = (float*)   alloc((size_t)NN * 4);
  float*    rowA2 = (float*)   alloc((size_t)NN * 4);
  float*    dis   = (float*)   alloc((size_t)NN * 4);

  // Build adjacency + row degrees.
  zero_f32<<<4096, 256, 0, stream>>>(A32, (size_t)NN * NN);
  zero_f32<<<16,   256, 0, stream>>>(rowA, (size_t)NN);
  scatter_edges<<<EE / 256, 256, 0, stream>>>(ei, A32, rowA, EE, NN);
  convert_A<<<(NN * (NN / 256)), 256, 0, stream>>>(A32, Ah, Ath, NN);

  // A2 = A @ A  (f16 inputs exact: small integer counts; f32 accumulate).
  wmma_gemm<<<dim3(NN / BN, NN / BM), 256, 0, stream>>>(Ah, Ath, A2, NN, NN, NN);
  rowsum_f32<<<NN, 256, 0, stream>>>(A2, rowA2, NN);

  // ----- layer 1 -----
  compute_dis<<<16, 256, 0, stream>>>(w1, rowA, rowA2, dis, NN);
  build_P<<<(NN * (NN / 256)), 256, 0, stream>>>(w1, A32, A2, Ph, (size_t)NN * NN);
  build_xst<<<(FP * NN) / 256, 256, 0, stream>>>(x, dis, xt, NN, FF, FP);
  wmma_gemm<<<dim3(FP / BN, NN / BM), 256, 0, stream>>>(Ph, xt, y, NN, FP, NN);
  linear_comb<<<(NN * HH) / 256, 256, 0, stream>>>(w1, dis, x, y, W1, b1, h,
                                                   NN, FF, FP, HH, /*relu=*/1);

  // ----- layer 2 -----
  compute_dis<<<16, 256, 0, stream>>>(w2, rowA, rowA2, dis, NN);
  build_P<<<(NN * (NN / 256)), 256, 0, stream>>>(w2, A32, A2, Ph, (size_t)NN * NN);
  build_xst<<<(HH * NN) / 256, 256, 0, stream>>>(h, dis, xt, NN, HH, HH);
  wmma_gemm<<<dim3(HH / BN, NN / BM), 256, 0, stream>>>(Ph, xt, y, NN, HH, NN);
  linear_comb<<<(NN * CC) / 256, 256, 0, stream>>>(w2, dis, h, y, W2, b2, out,
                                                   NN, HH, HH, CC, /*relu=*/0);

  logsoftmax_rows<<<(NN + 255) / 256, 256, 0, stream>>>(out, NN, CC);
}
